// Scale_Dot_Attention_57836029608017
// MI455X (gfx1250) — matrix operand order repaired and checked by reading, not compiled, not run on hardware
//
#include <hip/hip_runtime.h>
#include <hip/hip_bf16.h>
#include <math.h>

// ---------------------------------------------------------------------------
// Scaled-dot-product attention for MI455X (gfx1250, wave32, WMMA).
//   B=4, S=2048, D=1024, fp32 in/out, bf16 WMMA internally (f32 accum).
// Pipeline:
//   1) qkv_proj_kernel x3 : q/k/v = x @ W + b   (bf16 out into d_ws)
//   2) flash_attn_kernel  : streaming softmax(q k^T / 32) v  (fp32 out)
//      K tiles stream HBM/L2 -> LDS via GLOBAL_LOAD_ASYNC_TO_LDS_B128
//      (ASYNCcnt), double buffered so copies overlap WMMA compute.
// Workspace: 3 * 8192*1024 bf16 = 48 MB.
// ---------------------------------------------------------------------------

typedef __attribute__((ext_vector_type(16))) __bf16 v16bf;
typedef __attribute__((ext_vector_type(8)))  float  v8f;

#define WMMA_BF16(A, B, C) \
    __builtin_amdgcn_wmma_f32_16x16x32_bf16(false, (A), false, (B), (short)0, (C), false, false)

constexpr int Bb = 4, Ss = 2048, Dd = 1024;
constexpr int Mm = Bb * Ss;            // 8192 GEMM rows

// ---- CDNA5 async global->LDS copy (VGPR-bypassing, tracked by ASYNCcnt) ----
__device__ __forceinline__ void async_copy_b128(void* lds_ptr, const void* gptr) {
    // generic LDS pointer keeps the group-segment byte offset in its low 32 bits
    unsigned loff = (unsigned)(uintptr_t)lds_ptr;
    asm volatile("global_load_async_to_lds_b128 %0, %1, off"
                 :: "v"(loff), "v"(gptr) : "memory");
}
__device__ __forceinline__ void wait_async0() {
    asm volatile("s_wait_asynccnt 0x0" ::: "memory");
}

// =====================  Kernel 1: GEMM projection  =========================
// y[m,n] = sum_d x[m,d] * W[d,n] + bias[n],  M=8192, N=K=1024, bf16 output.
// Tile 128x128, 256 threads = 8 waves in 4(M)x2(N) grid; wave = 32x64 patch.
// X tile row-major [m][k]; W tile TRANSPOSED in LDS [n][k] so both A and B
// fragments are contiguous 16B ds_load_b128 pairs. Rows padded to 40 elems
// (80 B = 20 dwords) to spread the 16 fragment lanes across the 64 banks.
__global__ __launch_bounds__(256)
void qkv_proj_kernel(const float* __restrict__ x,
                     const float* __restrict__ W,
                     const float* __restrict__ bias,
                     __bf16* __restrict__ out)
{
    constexpr int N = Dd, K = Dd;
    constexpr int TP = 40;                         // padded tile row (bf16)
    __shared__ __align__(16) __bf16 Xs[128][TP];   // [m][k]  10 KB
    __shared__ __align__(16) __bf16 WsT[128][TP];  // [n][k]  10 KB

    const int tid  = threadIdx.x;
    const int wave = tid >> 5;
    const int lane = tid & 31;
    const int half = lane >> 4;
    const int l16  = lane & 15;
    const int wm   = wave >> 1;   // 0..3  (M group of 32 rows)
    const int wn   = wave & 1;    // 0..1  (N group of 64 cols)

    const int m0 = blockIdx.x * 128;
    const int n0 = blockIdx.y * 128;

    v8f acc[2][4] = {};

    for (int k0 = 0; k0 < K; k0 += 32) {
        // ---- stage x tile (128x32 f32 -> bf16, row-major) ----
        {
            const int c = (tid & 7) * 4;
            const int r = tid >> 3;      // 0..31
#pragma unroll
            for (int rr = 0; rr < 4; ++rr) {
                const int row = r + rr * 32;
                const float4 fv = *(const float4*)(x + (size_t)(m0 + row) * K + k0 + c);
                Xs[row][c + 0] = (__bf16)fv.x;
                Xs[row][c + 1] = (__bf16)fv.y;
                Xs[row][c + 2] = (__bf16)fv.z;
                Xs[row][c + 3] = (__bf16)fv.w;
            }
        }
        // ---- stage W tile (32x128 f32 -> bf16, transposed to [n][k]) ----
        {
            const int c = (tid & 31) * 4;  // n offset 0..124
            const int r = tid >> 5;        // k 0..7
#pragma unroll
            for (int rr = 0; rr < 4; ++rr) {
                const int row = r + rr * 8;
                const float4 fv = *(const float4*)(W + (size_t)(k0 + row) * N + n0 + c);
                WsT[c + 0][row] = (__bf16)fv.x;
                WsT[c + 1][row] = (__bf16)fv.y;
                WsT[c + 2][row] = (__bf16)fv.z;
                WsT[c + 3][row] = (__bf16)fv.w;
            }
        }
        __syncthreads();

        // ---- fragments (contiguous b128 LDS loads) + WMMA ----
        v16bf afrag[2];
#pragma unroll
        for (int i = 0; i < 2; ++i) {
            const int row = wm * 32 + i * 16 + l16;
#pragma unroll
            for (int e = 0; e < 8; ++e) {
                afrag[i][e]     = Xs[row][half * 8 + e];        // K = 8*half + e
                afrag[i][8 + e] = Xs[row][16 + half * 8 + e];   // K = 16 + 8*half + e
            }
        }
        v16bf bfrag[4];
#pragma unroll
        for (int j = 0; j < 4; ++j) {
            const int col = wn * 64 + j * 16 + l16;
#pragma unroll
            for (int e = 0; e < 8; ++e) {
                bfrag[j][e]     = WsT[col][half * 8 + e];       // K = 8*half + e
                bfrag[j][8 + e] = WsT[col][16 + half * 8 + e];  // K = 16 + 8*half + e
            }
        }
#pragma unroll
        for (int i = 0; i < 2; ++i)
#pragma unroll
            for (int j = 0; j < 4; ++j)
                acc[i][j] = WMMA_BF16(afrag[i], bfrag[j], acc[i][j]);
        __syncthreads();
    }

    // ---- bias + bf16 store (C layout: VGPR r -> row r + 8*half, col l16) ----
#pragma unroll
    for (int j = 0; j < 4; ++j) {
        const int col = n0 + wn * 64 + j * 16 + l16;
        const float bb = bias[col];
        __bf16* op = out + (size_t)(m0 + wm * 32 + 8 * half) * N + col;
#pragma unroll
        for (int i = 0; i < 2; ++i)
#pragma unroll
            for (int r = 0; r < 8; ++r)
                op[(size_t)(i * 16 + r) * N] = (__bf16)(acc[i][j][r] + bb);
    }
}

// =====================  Kernel 2: flash attention  =========================
// Per block: one batch b, 32 query rows. 512 threads = 16 waves, 2(row)x8(col).
// Streaming over 128-key blocks; online softmax in f32; P@V via WMMA.
// K is streamed into double-buffered LDS with async copies (ASYNCcnt).
__global__ __launch_bounds__(512)
void flash_attn_kernel(const __bf16* __restrict__ q,
                       const __bf16* __restrict__ k,
                       const __bf16* __restrict__ v,
                       float* __restrict__ out)
{
    constexpr int S = Ss, D = Dd;
    constexpr int Br = 32, Bc = 128, DC = 64;
    constexpr int NC = D / DC;                       // 16 d-chunks
    constexpr int QP = 1032, KP = 72, PP = 136;      // padded rows (bank spread)
    __shared__ __align__(16) __bf16 Qs[Br][QP];      // ~66 KB
    __shared__ __align__(16) __bf16 Ks[2][Bc][KP];   // ~36 KB (double buffer)
    __shared__ __align__(16) __bf16 Ps[Br][PP];      // ~8.5 KB
    __shared__ float redmax[2][16][8];
    __shared__ float redsum[2][16][8];

    const int b    = blockIdx.y;
    const int q0   = blockIdx.x * Br;
    const int tid  = threadIdx.x;
    const int wave = tid >> 5;
    const int lane = tid & 31;
    const int half = lane >> 4;
    const int l16  = lane & 15;
    const int rg   = wave >> 3;   // 0..1 : 16-row group
    const int cg   = wave & 7;    // 0..7 : 16-col (scores) / 128-col (O) group

    const size_t base = (size_t)b * S * D;
    const float  scale = 0.03125f;  // 1/sqrt(1024)

    // stage one 128x64 K chunk into Ks[buf] (2 async b128 copies per thread)
    auto stage_k = [&](int buf, const __bf16* kblk, int d0) {
        for (int i = tid; i < 1024; i += 512) {
            const int kr = i >> 3, kc = (i & 7) * 8;
            async_copy_b128(&Ks[buf][kr][kc], kblk + (size_t)kr * D + d0 + kc);
        }
    };

    // stage Q block (32 x 1024 bf16) + first K chunk, all async, one drain
    {
        const __bf16* src = q + base + (size_t)q0 * D;
        for (int i = tid; i < 4096; i += 512) {
            const int row = i >> 7;
            const int c   = (i & 127) * 8;
            async_copy_b128(&Qs[row][c], src + (size_t)row * D + c);
        }
        stage_k(0, k + base, 0);
        wait_async0();
        __syncthreads();
    }

    v8f   oacc[8] = {};
    float m_i[8], l_i[8];
#pragma unroll
    for (int r = 0; r < 8; ++r) { m_i[r] = -1.0e30f; l_i[r] = 0.0f; }

    const int nKB = S / Bc;  // 16
    for (int kb = 0; kb < nKB; ++kb) {
        const __bf16* kblk = k + base + (size_t)(kb * Bc) * D;
        // -------- Phase 1: scores tile [16 x 16] over full D --------
        v8f sacc = {};
        for (int c0 = 0; c0 < NC; ++c0) {
            const int cur = c0 & 1;
            // kick off the next chunk into the other buffer (overlaps WMMAs)
            if (c0 + 1 < NC)            stage_k(cur ^ 1, kblk, (c0 + 1) * DC);
            else if (kb + 1 < nKB)      stage_k(cur ^ 1, kblk + (size_t)Bc * D, 0);

            const int d0 = c0 * DC;
#pragma unroll
            for (int kk = 0; kk < 2; ++kk) {
                const int dk = d0 + kk * 32;
                v16bf af, bfK;
                const int row = rg * 16 + l16;
#pragma unroll
                for (int e = 0; e < 8; ++e) {
                    af[e]     = Qs[row][dk + half * 8 + e];
                    af[8 + e] = Qs[row][dk + 16 + half * 8 + e];
                }
                const int keyrow = cg * 16 + l16;   // B = K^T : B[d][n] = Ks[n][d]
#pragma unroll
                for (int e = 0; e < 8; ++e) {
                    bfK[e]     = Ks[cur][keyrow][kk * 32 + half * 8 + e];
                    bfK[8 + e] = Ks[cur][keyrow][kk * 32 + 16 + half * 8 + e];
                }
                sacc = WMMA_BF16(af, bfK, sacc);
            }
            wait_async0();     // just-issued copies complete during the WMMAs
            __syncthreads();   // hand buffers over
        }
#pragma unroll
        for (int r = 0; r < 8; ++r) sacc[r] *= scale;

        // -------- online softmax --------
        float tmax[8];
#pragma unroll
        for (int r = 0; r < 8; ++r) {
            float mv = sacc[r];
#pragma unroll
            for (int s = 1; s < 16; s <<= 1) mv = fmaxf(mv, __shfl_xor(mv, s, 32));
            tmax[r] = mv;
        }
        if (l16 == 0) {
#pragma unroll
            for (int r = 0; r < 8; ++r) redmax[rg][r + 8 * half][cg] = tmax[r];
        }
        __syncthreads();
        float alpha[8];
#pragma unroll
        for (int r = 0; r < 8; ++r) {
            float bm = -1.0e30f;
#pragma unroll
            for (int c = 0; c < 8; ++c) bm = fmaxf(bm, redmax[rg][r + 8 * half][c]);
            const float mnew = fmaxf(m_i[r], bm);
            alpha[r] = __expf(m_i[r] - mnew);
            m_i[r]   = mnew;
        }
        // P = exp(S - m), write bf16 tile, accumulate row sums
#pragma unroll
        for (int r = 0; r < 8; ++r) {
            const float p = __expf(sacc[r] - m_i[r]);
            sacc[r] = p;
            Ps[rg * 16 + r + 8 * half][cg * 16 + l16] = (__bf16)p;
            float sv = p;
#pragma unroll
            for (int s = 1; s < 16; s <<= 1) sv += __shfl_xor(sv, s, 32);
            if (l16 == 0) redsum[rg][r + 8 * half][cg] = sv;
        }
        __syncthreads();   // Ps + redsum visible to everyone
#pragma unroll
        for (int r = 0; r < 8; ++r) {
            float bs = 0.0f;
#pragma unroll
            for (int c = 0; c < 8; ++c) bs += redsum[rg][r + 8 * half][c];
            l_i[r] = l_i[r] * alpha[r] + bs;
        }
        // rescale O accumulators (row mapping identical to C layout)
#pragma unroll
        for (int j2 = 0; j2 < 8; ++j2)
#pragma unroll
            for (int r = 0; r < 8; ++r) oacc[j2][r] *= alpha[r];

        // -------- Phase 2: O += P @ V  (wave: 16 rows x 128 cols) --------
        v16bf pf[4];
#pragma unroll
        for (int kk = 0; kk < 4; ++kk) {
            const int prow = rg * 16 + l16;
            const int kbse = kk * 32;
#pragma unroll
            for (int e = 0; e < 8; ++e) {
                pf[kk][e]     = Ps[prow][kbse + half * 8 + e];
                pf[kk][8 + e] = Ps[prow][kbse + 16 + half * 8 + e];
            }
        }
#pragma unroll
        for (int j2 = 0; j2 < 8; ++j2) {
            const int ncol = cg * 128 + j2 * 16 + l16;
#pragma unroll
            for (int kk = 0; kk < 4; ++kk) {
                // base pointer once; e*D folds into const instruction offsets
                const __bf16* vp =
                    v + base + (size_t)(kb * Bc + kk * 32 + half * 8) * D + ncol;
                v16bf bfV;
#pragma unroll
                for (int e = 0; e < 16; ++e) bfV[e] = vp[(size_t)(e < 8 ? e : e + 8) * D];
                oacc[j2] = WMMA_BF16(pf[kk], bfV, oacc[j2]);
            }
        }
    }

    // -------- finalize: O / l, fp32 store --------
    float inv_l[8];
#pragma unroll
    for (int r = 0; r < 8; ++r) inv_l[r] = 1.0f / l_i[r];
#pragma unroll
    for (int j2 = 0; j2 < 8; ++j2) {
        const int ncol = cg * 128 + j2 * 16 + l16;
        float* op = out + base + (size_t)(q0 + rg * 16 + 8 * half) * D + ncol;
#pragma unroll
        for (int r = 0; r < 8; ++r)
            op[(size_t)r * D] = oacc[j2][r] * inv_l[r];
    }
}

// ==========================  host launcher  ================================
extern "C" void kernel_launch(void* const* d_in, const int* in_sizes, int n_in,
                              void* d_out, int out_size, void* d_ws, size_t ws_size,
                              hipStream_t stream) {
    const float* x  = (const float*)d_in[0];
    const float* Wq = (const float*)d_in[1];
    const float* bq = (const float*)d_in[2];
    const float* Wk = (const float*)d_in[3];
    const float* bk = (const float*)d_in[4];
    const float* Wv = (const float*)d_in[5];
    const float* bv = (const float*)d_in[6];
    float* out = (float*)d_out;

    __bf16* qb = (__bf16*)d_ws;                         // 16 MB
    __bf16* kb = qb + (size_t)Mm * Dd;                  // 16 MB
    __bf16* vb = kb + (size_t)Mm * Dd;                  // 16 MB  (48 MB total)

    dim3 gProj(Mm / 128, Dd / 128);   // 64 x 8
    qkv_proj_kernel<<<gProj, 256, 0, stream>>>(x, Wq, bq, qb);
    qkv_proj_kernel<<<gProj, 256, 0, stream>>>(x, Wk, bk, kb);
    qkv_proj_kernel<<<gProj, 256, 0, stream>>>(x, Wv, bv, vb);

    dim3 gAttn(Ss / 32, Bb);          // 64 x 4
    flash_attn_kernel<<<gAttn, 512, 0, stream>>>(qb, kb, vb, out);
}
